// PerformerOffsetAttention_43413529428105
// MI455X (gfx1250) — matrix operand order repaired and checked by reading, not compiled, not run on hardware
//
#include <hip/hip_runtime.h>
#include <hip/hip_bf16.h>
#include <math.h>

#define BATCH 4
#define CH    256
#define NPTS  8192
#define DA    64
#define NB    64
#define RSM   128   // 2*NB
#define RG    16
#define RTOT  2048  // RSM*RG

typedef __attribute__((ext_vector_type(16))) __bf16 v16bf;
typedef __attribute__((ext_vector_type(8)))  float  v8f;
typedef __attribute__((ext_vector_type(8)))  unsigned v8u;

static __device__ __forceinline__ unsigned short f2bfbits(float f) {
  unsigned u = __builtin_bit_cast(unsigned, f);
  u += 0x7FFFu + ((u >> 16) & 1u);          // round-to-nearest-even
  return (unsigned short)(u >> 16);
}
static __device__ __forceinline__ float bfbits2f(unsigned short s) {
  return __builtin_bit_cast(float, ((unsigned)s) << 16);
}
static __device__ __forceinline__ float blo(unsigned u) {
  return __builtin_bit_cast(float, u << 16);
}
static __device__ __forceinline__ float bhi(unsigned u) {
  return __builtin_bit_cast(float, u & 0xFFFF0000u);
}
static __device__ __forceinline__ unsigned packbf(float lo, float hi) {
  return (unsigned)f2bfbits(lo) | ((unsigned)f2bfbits(hi) << 16);
}

// ---- gfx1250 async global->LDS path (ASYNCcnt) ------------------------------
static __device__ __forceinline__ unsigned lds_off(const void* p) {
  return (unsigned)(unsigned long long)(__attribute__((address_space(3))) const void*)p;
}
static __device__ __forceinline__ void async_ld_b128(unsigned ldsoff, const void* g) {
  unsigned long long ga = (unsigned long long)g;
  asm volatile("global_load_async_to_lds_b128 %0, %1, off" :: "v"(ldsoff), "v"(ga) : "memory");
}
static __device__ __forceinline__ void wait_async0() { asm volatile("s_wait_asynccnt 0x0" ::: "memory"); }
static __device__ __forceinline__ void wait_async1() { asm volatile("s_wait_asynccnt 0x1" ::: "memory"); }
static __device__ __forceinline__ void wait_async2() { asm volatile("s_wait_asynccnt 0x2" ::: "memory"); }

// ---------------------------------------------------------------------------
// K0: convert Wv, Wp to bf16 once per launch
// ---------------------------------------------------------------------------
__global__ __launch_bounds__(256)
void wcvt_kernel(const float* __restrict__ Wv, const float* __restrict__ Wp,
                 unsigned short* __restrict__ wvb, unsigned short* __restrict__ wpb)
{
  int i = blockIdx.x * 256 + threadIdx.x;   // 65536
  wvb[i] = f2bfbits(Wv[i]);
  wpb[i] = f2bfbits(Wp[i]);
}

// ---------------------------------------------------------------------------
// K1: q/k projections + FAVOR+ features + geo RFF phi + bf16 x transpose.
// Outputs: qp[B][N][128] (row), kpT[B][128][N], phi[B][N][16], phiT[B][16][N],
//          xb[B][N][256] (point-major bf16)
// block = 256 threads, 32 points per block
// ---------------------------------------------------------------------------
__global__ __launch_bounds__(256)
void features_kernel(const float* __restrict__ x, const float* __restrict__ pos,
                     const float* __restrict__ Wq, const float* __restrict__ Wk,
                     const float* __restrict__ Pm,
                     const float* __restrict__ Wg1, const float* __restrict__ bg1,
                     const float* __restrict__ Wg2, const float* __restrict__ bg2,
                     const float* __restrict__ log_tau, const float* __restrict__ omega,
                     unsigned short* __restrict__ qp, unsigned short* __restrict__ kpT,
                     unsigned short* __restrict__ phi, unsigned short* __restrict__ phiT,
                     unsigned short* __restrict__ xb)
{
  __shared__ float smem[12288];               // 48 KB
  float* Xs = smem;                           // [256][32]
  float* QK = smem + 8192;                    // [128][32]
  float* PR = smem;                           // [128][32] reuses Xs region

  const int tid = threadIdx.x;
  int blk = blockIdx.x;
  const int b  = blk / (NPTS / 32);
  const int n0 = (blk % (NPTS / 32)) * 32;

  for (int i = tid; i < 256 * 32; i += 256) {
    int c = i >> 5, p = i & 31;
    Xs[i] = x[((size_t)b * CH + c) * NPTS + n0 + p];
  }
  __syncthreads();

  // point-major bf16 copy of x (for v-GEMM B fragments)
  for (int i = tid; i < 256 * 32; i += 256) {
    int p = i >> 8, c = i & 255;
    xb[((size_t)b * NPTS + n0 + p) * CH + c] = f2bfbits(Xs[c * 32 + p]);
  }

  const float qscale = 0.35355339059327373f;  // 64^-0.25
  for (int t = tid; t < 128 * 32; t += 256) {
    int o = t >> 5, p = t & 31;
    const float* W = (o < 64) ? (Wq + (size_t)o * CH) : (Wk + (size_t)(o - 64) * CH);
    float acc = 0.f;
    for (int c = 0; c < CH; ++c) acc += W[c] * Xs[c * 32 + p];
    QK[t] = acc * qscale;
  }
  __syncthreads();

  for (int t = tid; t < 128 * 32; t += 256) {
    int j = t >> 5, p = t & 31;
    int base = (j < 64) ? 0 : 64;
    int jj = j & 63;
    float acc = 0.f;
    for (int i = 0; i < 64; ++i) acc += QK[(base + i) * 32 + p] * Pm[i * 64 + jj];
    PR[t] = acc;
  }
  __syncthreads();

  const float sfeat = 0.08838834764831845f;   // (2*NB)^-0.5
  if (tid < 64) {
    int p = tid & 31, isK = tid >> 5;
    int n = n0 + p;
    int base = isK * 64;
    float mx = 0.f;
    for (int j = 0; j < 64; ++j) mx = fmaxf(mx, fabsf(PR[(base + j) * 32 + p]));
    if (!isK) {
      unsigned short* dst = qp + ((size_t)b * NPTS + n) * RSM;
      for (int j = 0; j < 64; ++j) {
        float pr = PR[(base + j) * 32 + p];
        dst[j]      = f2bfbits(__expf(pr  - mx) * sfeat + 1e-6f);
        dst[64 + j] = f2bfbits(__expf(-pr - mx) * sfeat + 1e-6f);
      }
    } else {
      for (int j = 0; j < 64; ++j) {
        float pr = PR[(base + j) * 32 + p];
        kpT[((size_t)b * RSM + j)      * NPTS + n] = f2bfbits(__expf(pr  - mx) * sfeat + 1e-6f);
        kpT[((size_t)b * RSM + 64 + j) * NPTS + n] = f2bfbits(__expf(-pr - mx) * sfeat + 1e-6f);
      }
    }
  } else if (tid < 96) {
    int p = tid - 64;
    int n = n0 + p;
    size_t pb = ((size_t)b * NPTS + n) * 3;
    float px = pos[pb], py = pos[pb + 1], pz = pos[pb + 2];
    float tau = log1pf(__expf(log_tau[0])) + 1e-6f;
    float s2t = sqrtf(2.f * tau);
    float g1[16], z[16];
    for (int o = 0; o < 16; ++o) {
      float a = Wg1[o * 3] * px + Wg1[o * 3 + 1] * py + Wg1[o * 3 + 2] * pz + bg1[o];
      g1[o] = fmaxf(a, 0.f);
    }
    float znorm = 0.f;
    for (int o = 0; o < 16; ++o) {
      float a = bg2[o];
      for (int i = 0; i < 16; ++i) a += Wg2[o * 16 + i] * g1[i];
      z[o] = s2t * a;
      znorm += z[o] * z[o];
    }
    znorm *= 0.5f;
    unsigned short* dst = phi + ((size_t)b * NPTS + n) * RG;
    for (int j = 0; j < 16; ++j) {
      float zp = 0.f;
      for (int i = 0; i < 16; ++i) zp += z[i] * omega[i * 16 + j];
      unsigned short v = f2bfbits(__expf(zp - znorm) * 0.25f);   // 16^-0.5
      dst[j] = v;
      phiT[((size_t)b * RG + j) * NPTS + n] = v;
    }
  }
}

// ---------------------------------------------------------------------------
// K2: v = Wv @ x -> vpi[B][N/2][CH][2] (pair-interleaved bf16). 1 wave / tile.
// ---------------------------------------------------------------------------
__global__ __launch_bounds__(32)
void v_gemm_kernel(const unsigned short* __restrict__ xb, const unsigned short* __restrict__ wvb,
                   unsigned short* __restrict__ vpi)
{
  const int lane = threadIdx.x;
  int blk = blockIdx.x;
  const int nt = blk % (NPTS / 16); blk /= (NPTS / 16);
  const int cm = blk % (CH / 16);   blk /= (CH / 16);
  const int b  = blk;
  const int c0 = cm * 16, n0 = nt * 16;
  const int lm  = lane & 15;
  const int hib = (lane >> 4) & 1;
  const int kbase = hib ? 8 : 0;
  const int koffB = hib ? 8 : 0;
  v8f acc = {};
  const unsigned short* Arow = wvb + (size_t)(c0 + lm) * CH;
  const unsigned short* Bcol = xb + ((size_t)b * NPTS + n0 + lm) * CH;
  for (int k0 = 0; k0 < CH; k0 += 32) {
    v8u ua, ub;
    uint4 a0 = *(const uint4*)(Arow + k0 + kbase);
    uint4 a1 = *(const uint4*)(Arow + k0 + 16 + kbase);
    ua[0] = a0.x; ua[1] = a0.y; ua[2] = a0.z; ua[3] = a0.w;
    ua[4] = a1.x; ua[5] = a1.y; ua[6] = a1.z; ua[7] = a1.w;
    uint4 b0 = *(const uint4*)(Bcol + k0 + koffB);
    uint4 b1 = *(const uint4*)(Bcol + k0 + koffB + 16);
    ub[0] = b0.x; ub[1] = b0.y; ub[2] = b0.z; ub[3] = b0.w;
    ub[4] = b1.x; ub[5] = b1.y; ub[6] = b1.z; ub[7] = b1.w;
    acc = __builtin_amdgcn_wmma_f32_16x16x32_bf16(
        false, __builtin_bit_cast(v16bf, ua), false, __builtin_bit_cast(v16bf, ub),
        (short)0, acc, false, false);
  }
  // D: rows c = c0 + j + 8*hib, col n = n0 + lm  -> pair-interleaved on n
  const int n = n0 + lm;
  for (int j = 0; j < 8; ++j) {
    int c = c0 + 8 * hib + j;
    vpi[((size_t)b * (NPTS/2) + (n >> 1)) * (CH*2) + c * 2 + (n & 1)] = f2bfbits(acc[j]);
  }
}

// ---------------------------------------------------------------------------
// K3: ksum[b][r] = sum_n Phi_k[n][r]   (coalesced via kpT / phiT)
// ---------------------------------------------------------------------------
__global__ __launch_bounds__(256)
void ksum_kernel(const unsigned short* __restrict__ kpT, const unsigned short* __restrict__ phiT,
                 float* __restrict__ ksum)
{
  __shared__ float sred[256];
  int blk = blockIdx.x;
  const int smv = blk % RSM;
  const int b   = blk / RSM;
  const int tid = threadIdx.x;
  const unsigned short* kprow = kpT + ((size_t)b * RSM + smv) * NPTS;
  float acc[16];
  for (int j = 0; j < 16; ++j) acc[j] = 0.f;
  for (int n = tid; n < NPTS; n += 256) {
    float kv = bfbits2f(kprow[n]);
    for (int j = 0; j < 16; ++j)
      acc[j] += kv * bfbits2f(phiT[((size_t)b * RG + j) * NPTS + n]);
  }
  for (int j = 0; j < 16; ++j) {
    sred[tid] = acc[j];
    __syncthreads();
    for (int s = 128; s > 0; s >>= 1) {
      if (tid < s) sred[tid] += sred[tid + s];
      __syncthreads();
    }
    if (tid == 0) ksum[(size_t)b * RTOT + smv * RG + j] = sred[0];
    __syncthreads();
  }
}

// ---------------------------------------------------------------------------
// K4: kv = Phi_k^T @ v (Phi on the fly). 4 waves / block share an async-staged
//     double-buffered LDS v-tile (32 pts x 64 ch). Output pair-interleaved.
// ---------------------------------------------------------------------------
__global__ __launch_bounds__(128)
void kv_gemm_kernel(const unsigned short* __restrict__ kpT, const unsigned short* __restrict__ phiT,
                    const unsigned short* __restrict__ vpi, unsigned short* __restrict__ kvpi)
{
  __shared__ unsigned short vtile[2][32 * 64];   // 2 x 4KB
  const int tid  = threadIdx.x;
  const int lane = tid & 31;
  const int w    = tid >> 5;
  int blk = blockIdx.x;                          // B * RSM * (CH/64)
  const int cb  = blk % (CH / 64); blk /= (CH / 64);
  const int smv = blk % RSM;       blk /= RSM;
  const int b   = blk;
  const int cb0 = cb * 64;
  const int lm  = lane & 15;
  const int hib = (lane >> 4) & 1;
  const int kbase = hib ? 8 : 0;
  const int cw  = w * 16 + lm;                   // column within block tile

  const unsigned short* kprow = kpT  + ((size_t)b * RSM + smv) * NPTS;
  const unsigned short* phrow = phiT + ((size_t)b * RG + lm) * NPTS;
  v8f acc = {};

  // stage helper (lambda-free): issues 2 async b128 per thread
  const int NS = NPTS / 32;
  {
    // prologue stage: buffer 0, n0 = 0
    for (int t = 0; t < 2; ++t) {
      int q = tid + t * 128;
      int row = q >> 4, ch = q & 15;
      const unsigned short* g = vpi + ((size_t)b * (NPTS/2) + row) * (CH*2) + cb0 * 2 + ch * 8;
      async_ld_b128(lds_off(&vtile[0][row * 128 + ch * 8]), g);
    }
  }
  for (int it = 0; it < NS; ++it) {
    const int buf = it & 1;
    if (it + 1 < NS) {
      int np0n = (it + 1) * 16;
      for (int t = 0; t < 2; ++t) {
        int q = tid + t * 128;
        int row = q >> 4, ch = q & 15;
        const unsigned short* g = vpi + ((size_t)b * (NPTS/2) + np0n + row) * (CH*2) + cb0 * 2 + ch * 8;
        async_ld_b128(lds_off(&vtile[buf ^ 1][row * 128 + ch * 8]), g);
      }
      wait_async2();
    } else {
      wait_async0();
    }
    __syncthreads();

    const int n0 = it * 32;
    // A: Phi_k^T tile (16 geo x 32 n) built from kpT (wave-uniform) and phiT
    uint4 kq0 = *(const uint4*)(kprow + n0 + kbase);
    uint4 kq1 = *(const uint4*)(kprow + n0 + 16 + kbase);
    uint4 pq0 = *(const uint4*)(phrow + n0 + kbase);
    uint4 pq1 = *(const uint4*)(phrow + n0 + 16 + kbase);
    unsigned ka[8] = {kq0.x, kq0.y, kq0.z, kq0.w, kq1.x, kq1.y, kq1.z, kq1.w};
    unsigned pa[8] = {pq0.x, pq0.y, pq0.z, pq0.w, pq1.x, pq1.y, pq1.z, pq1.w};
    v8u ua, ub;
    for (int j = 0; j < 8; ++j)
      ua[j] = packbf(blo(ka[j]) * blo(pa[j]), bhi(ka[j]) * bhi(pa[j]));
    // B: v tile from LDS (dword = K-pair)
    const unsigned* vt32 = (const unsigned*)vtile[buf];
    const int np0 = hib ? 4 : 0;
    for (int j = 0; j < 8; ++j) ub[j] = vt32[(np0 + (j & 3) + (j >> 2) * 8) * 64 + cw];
    acc = __builtin_amdgcn_wmma_f32_16x16x32_bf16(
        false, __builtin_bit_cast(v16bf, ua), false, __builtin_bit_cast(v16bf, ub),
        (short)0, acc, false, false);
    __syncthreads();
  }
  // D: r = smv*16 + 8*hib + j, c = cb0 + cw ; pack row pairs -> dwords
  unsigned* kv32 = (unsigned*)kvpi;
  for (int t = 0; t < 4; ++t) {
    unsigned d = packbf(acc[2 * t], acc[2 * t + 1]);
    kv32[((size_t)b * (RTOT/2) + smv * 8 + 4 * hib + t) * CH + cb0 + cw] = d;
  }
}

// ---------------------------------------------------------------------------
// K5: denom[b][n] = Phi_q[n] . ksum[b] + eps  (ksum staged in LDS)
// ---------------------------------------------------------------------------
__global__ __launch_bounds__(256)
void denom_kernel(const unsigned short* __restrict__ qp, const unsigned short* __restrict__ phi,
                  const float* __restrict__ ksum, float* __restrict__ denom)
{
  __shared__ float ks[RTOT];
  int blk = blockIdx.x;
  const int b  = blk / (NPTS / 256);
  const int n0 = (blk % (NPTS / 256)) * 256;
  const int tid = threadIdx.x;
  for (int i = tid; i < RTOT; i += 256) ks[i] = ksum[(size_t)b * RTOT + i];
  __syncthreads();
  const int n = n0 + tid;
  const unsigned short* qr = qp  + ((size_t)b * NPTS + n) * RSM;
  const unsigned short* pr = phi + ((size_t)b * NPTS + n) * RG;
  float ph[16];
  for (int j = 0; j < 16; ++j) ph[j] = bfbits2f(pr[j]);
  float acc = 0.f;
  for (int s = 0; s < RSM; ++s) {
    float q = bfbits2f(qr[s]);
    float d = 0.f;
    for (int j = 0; j < 16; ++j) d += ph[j] * ks[s * 16 + j];
    acc += q * d;
  }
  denom[(size_t)b * NPTS + n] = acc + 1e-6f;
}

// ---------------------------------------------------------------------------
// K6: fsa = (Phi_q @ kv)/denom ; y = x - fsa^T (bf16 point-major).
//     4 waves (4 n-tiles) share async-staged double-buffered LDS kv-tile.
// ---------------------------------------------------------------------------
__global__ __launch_bounds__(128)
void fsa_y_kernel(const unsigned short* __restrict__ qp, const unsigned short* __restrict__ phi,
                  const unsigned short* __restrict__ kvpi, const float* __restrict__ denom,
                  const float* __restrict__ x, unsigned short* __restrict__ ybf)
{
  __shared__ unsigned short ktile[2][16 * 32];   // 2 x 1KB
  const int tid  = threadIdx.x;
  const int lane = tid & 31;
  const int w    = tid >> 5;
  int blk = blockIdx.x;                          // B * (NPTS/64) * (CH/16)
  const int ct  = blk % (CH / 16);   blk /= (CH / 16);
  const int ntb = blk % (NPTS / 64); blk /= (NPTS / 64);
  const int b   = blk;
  const int c0  = ct * 16;
  const int n0  = ntb * 64 + w * 16;
  const int lm  = lane & 15;
  const int hib = (lane >> 4) & 1;
  const int kbase = hib ? 8 : 0;

  const unsigned short* qr = qp + ((size_t)b * NPTS + n0 + lm) * RSM;
  const unsigned* pr32 = (const unsigned*)(phi + ((size_t)b * NPTS + n0 + lm) * RG);
  unsigned ph2[8];
  for (int j = 0; j < 8; ++j) ph2[j] = pr32[j];

  v8f acc = {};
  const int NS = RTOT / 32;
  if (tid < 64) {
    int row = tid >> 2, ch = tid & 3;
    const unsigned short* g = kvpi + ((size_t)b * (RTOT/2) + row) * (CH*2) + c0 * 2 + ch * 8;
    async_ld_b128(lds_off(&ktile[0][row * 32 + ch * 8]), g);
  }
  for (int it = 0; it < NS; ++it) {
    const int buf = it & 1;
    if (it + 1 < NS) {
      if (tid < 64) {
        int row = tid >> 2, ch = tid & 3;
        int np0n = (it + 1) * 16;
        const unsigned short* g = kvpi + ((size_t)b * (RTOT/2) + np0n + row) * (CH*2) + c0 * 2 + ch * 8;
        async_ld_b128(lds_off(&ktile[buf ^ 1][row * 32 + ch * 8]), g);
      }
      wait_async1();
    } else {
      wait_async0();
    }
    __syncthreads();

    const int r0 = it * 32;
    float q0 = bfbits2f(qr[(r0 >> 4)]);
    float q1 = bfbits2f(qr[(r0 >> 4) + 1]);
    v8u ua, ub;
    for (int j = 0; j < 4; ++j) {
      unsigned p2 = ph2[(kbase >> 1) + j];
      ua[j] = packbf(q0 * blo(p2), q0 * bhi(p2));
    }
    for (int j = 4; j < 8; ++j) {
      unsigned p2 = ph2[(kbase >> 1) + (j - 4)];
      ua[j] = packbf(q1 * blo(p2), q1 * bhi(p2));
    }
    const unsigned* t32 = (const unsigned*)ktile[buf];
    const int np0 = hib ? 4 : 0;
    for (int j = 0; j < 8; ++j) ub[j] = t32[(np0 + (j & 3) + (j >> 2) * 8) * 16 + lm];
    acc = __builtin_amdgcn_wmma_f32_16x16x32_bf16(
        false, __builtin_bit_cast(v16bf, ua), false, __builtin_bit_cast(v16bf, ub),
        (short)0, acc, false, false);
    __syncthreads();
  }
  // D: rows n = n0 + j + 8*hib, col c = c0 + lm
  for (int j = 0; j < 8; ++j) {
    int n = n0 + 8 * hib + j;
    float fsa = acc[j] / denom[(size_t)b * NPTS + n];
    float yv  = x[((size_t)b * CH + c0 + lm) * NPTS + n] - fsa;
    ybf[((size_t)b * NPTS + n) * CH + c0 + lm] = f2bfbits(yv);
  }
}

// ---------------------------------------------------------------------------
// K7: h = Wp @ y -> d_out (f32 [B][C][N])
// ---------------------------------------------------------------------------
__global__ __launch_bounds__(32)
void h_gemm_kernel(const unsigned short* __restrict__ ybf, const unsigned short* __restrict__ wpb,
                   float* __restrict__ hout)
{
  const int lane = threadIdx.x;
  int blk = blockIdx.x;
  const int nt = blk % (NPTS / 16); blk /= (NPTS / 16);
  const int cm = blk % (CH / 16);   blk /= (CH / 16);
  const int b  = blk;
  const int c0 = cm * 16, n0 = nt * 16;
  const int lm  = lane & 15;
  const int hib = (lane >> 4) & 1;
  const int kbase = hib ? 8 : 0;
  const int koffB = hib ? 8 : 0;
  v8f acc = {};
  const unsigned short* Arow = wpb + (size_t)(c0 + lm) * CH;
  const unsigned short* Bcol = ybf + ((size_t)b * NPTS + n0 + lm) * CH;
  for (int k0 = 0; k0 < CH; k0 += 32) {
    __builtin_prefetch(Bcol + k0 + 64, 0, 1);
    v8u ua, ub;
    uint4 a0 = *(const uint4*)(Arow + k0 + kbase);
    uint4 a1 = *(const uint4*)(Arow + k0 + 16 + kbase);
    ua[0] = a0.x; ua[1] = a0.y; ua[2] = a0.z; ua[3] = a0.w;
    ua[4] = a1.x; ua[5] = a1.y; ua[6] = a1.z; ua[7] = a1.w;
    uint4 b0 = *(const uint4*)(Bcol + k0 + koffB);
    uint4 b1 = *(const uint4*)(Bcol + k0 + koffB + 16);
    ub[0] = b0.x; ub[1] = b0.y; ub[2] = b0.z; ub[3] = b0.w;
    ub[4] = b1.x; ub[5] = b1.y; ub[6] = b1.z; ub[7] = b1.w;
    acc = __builtin_amdgcn_wmma_f32_16x16x32_bf16(
        false, __builtin_bit_cast(v16bf, ua), false, __builtin_bit_cast(v16bf, ub),
        (short)0, acc, false, false);
  }
  for (int j = 0; j < 8; ++j) {
    int c = c0 + 8 * hib + j;
    hout[((size_t)b * CH + c) * NPTS + n0 + lm] = acc[j];
  }
}

// ---------------------------------------------------------------------------
// K8: BN training statistics per channel
// ---------------------------------------------------------------------------
__global__ __launch_bounds__(256)
void bn_stats_kernel(const float* __restrict__ h, float* __restrict__ mean, float* __restrict__ rstd)
{
  __shared__ float s1[256], s2[256];
  const int c = blockIdx.x, tid = threadIdx.x;
  float a = 0.f, a2 = 0.f;
  for (int i = tid; i < BATCH * NPTS; i += 256) {
    int b = i / NPTS, n = i % NPTS;
    float v = h[((size_t)b * CH + c) * NPTS + n];
    a += v; a2 += v * v;
  }
  s1[tid] = a; s2[tid] = a2;
  __syncthreads();
  for (int s = 128; s > 0; s >>= 1) {
    if (tid < s) { s1[tid] += s1[tid + s]; s2[tid] += s2[tid + s]; }
    __syncthreads();
  }
  if (tid == 0) {
    const float inv = 1.f / (float)(BATCH * NPTS);
    float m = s1[0] * inv;
    float v = s2[0] * inv - m * m;
    mean[c] = m;
    rstd[c] = rsqrtf(v + 1e-5f);
  }
}

// ---------------------------------------------------------------------------
// K9: out = relu(BN(h)) + x   (in place on d_out)
// ---------------------------------------------------------------------------
__global__ __launch_bounds__(256)
void final_kernel(const float* __restrict__ x, const float* __restrict__ mean,
                  const float* __restrict__ rstd, const float* __restrict__ gamma,
                  const float* __restrict__ beta, float* __restrict__ out)
{
  size_t i = (size_t)blockIdx.x * 256 + threadIdx.x;
  int c = (int)((i / NPTS) % CH);
  float hv = out[i];
  float v = (hv - mean[c]) * rstd[c] * gamma[c] + beta[c];
  out[i] = fmaxf(v, 0.f) + x[i];
}

// ---------------------------------------------------------------------------
extern "C" void kernel_launch(void* const* d_in, const int* in_sizes, int n_in,
                              void* d_out, int out_size, void* d_ws, size_t ws_size,
                              hipStream_t stream)
{
  (void)in_sizes; (void)n_in; (void)out_size; (void)ws_size;
  const float* x    = (const float*)d_in[0];
  const float* pos  = (const float*)d_in[1];
  const float* Wq   = (const float*)d_in[2];
  const float* Wk   = (const float*)d_in[3];
  const float* Wv   = (const float*)d_in[4];
  const float* Pm   = (const float*)d_in[5];
  const float* Wg1  = (const float*)d_in[6];
  const float* bg1  = (const float*)d_in[7];
  const float* Wg2  = (const float*)d_in[8];
  const float* bg2  = (const float*)d_in[9];
  const float* ltau = (const float*)d_in[10];
  const float* om   = (const float*)d_in[11];
  const float* Wp   = (const float*)d_in[12];
  const float* gam  = (const float*)d_in[13];
  const float* bet  = (const float*)d_in[14];

  char* ws = (char*)d_ws;
  unsigned short* qp   = (unsigned short*)(ws + 0);          //  8 MB  [B][N][128]
  unsigned short* kpT  = (unsigned short*)(ws + 8388608);    //  8 MB  [B][128][N]
  unsigned short* phi  = (unsigned short*)(ws + 16777216);   //  1 MB  [B][N][16]
  unsigned short* phiT = (unsigned short*)(ws + 17825792);   //  1 MB  [B][16][N]
  unsigned short* vpi  = (unsigned short*)(ws + 18874368);   // 16 MB  [B][N/2][CH][2]
  unsigned short* kvpi = (unsigned short*)(ws + 35651584);   //  4 MB  [B][R/2][CH][2]
  float*          ksum = (float*)(ws + 39845888);            // 32 KB
  float*          den  = (float*)(ws + 39878656);            // 128 KB
  unsigned short* ybf  = (unsigned short*)(ws + 40009728);   // 16 MB  [B][N][CH]
  unsigned short* xb   = (unsigned short*)(ws + 56786944);   // 16 MB  [B][N][CH]
  unsigned short* wvb  = (unsigned short*)(ws + 73564160);   // 128 KB
  unsigned short* wpb  = (unsigned short*)(ws + 73695232);   // 128 KB
  float*          mean = (float*)(ws + 73826304);
  float*          rstd = (float*)(ws + 73827328);
  float* out = (float*)d_out;

  wcvt_kernel<<<256, 256, 0, stream>>>(Wv, Wp, wvb, wpb);
  features_kernel<<<BATCH * (NPTS / 32), 256, 0, stream>>>(
      x, pos, Wq, Wk, Pm, Wg1, bg1, Wg2, bg2, ltau, om, qp, kpT, phi, phiT, xb);
  v_gemm_kernel<<<BATCH * (CH / 16) * (NPTS / 16), 32, 0, stream>>>(xb, wvb, vpi);
  ksum_kernel<<<BATCH * RSM, 256, 0, stream>>>(kpT, phiT, ksum);
  kv_gemm_kernel<<<BATCH * RSM * (CH / 64), 128, 0, stream>>>(kpT, phiT, vpi, kvpi);
  denom_kernel<<<BATCH * (NPTS / 256), 256, 0, stream>>>(qp, phi, ksum, den);
  fsa_y_kernel<<<BATCH * (NPTS / 64) * (CH / 16), 128, 0, stream>>>(qp, phi, kvpi, den, x, ybf);
  h_gemm_kernel<<<BATCH * (CH / 16) * (NPTS / 16), 32, 0, stream>>>(ybf, wpb, out);
  bn_stats_kernel<<<CH, 256, 0, stream>>>(out, mean, rstd);
  final_kernel<<<(BATCH * CH * NPTS) / 256, 256, 0, stream>>>(x, mean, rstd, gam, bet, out);
}
